// MAN_9723805958786
// MI455X (gfx1250) — matrix operand order repaired and checked by reading, not compiled, not run on hardware
//
#include <hip/hip_runtime.h>
#include <hip/hip_fp16.h>
#include <math.h>

typedef __attribute__((ext_vector_type(16))) _Float16 v16h;
typedef __attribute__((ext_vector_type(8)))  _Float16 v8h;
typedef __attribute__((ext_vector_type(8)))  float    v8f;
typedef __attribute__((ext_vector_type(4)))  unsigned v4u;
typedef __attribute__((ext_vector_type(8)))  int      v8i_t;
typedef __attribute__((ext_vector_type(4)))  int      v4i_t;

#define WMMA16(a, b, c) \
  __builtin_amdgcn_wmma_f32_16x16x32_f16(false, (a), false, (b), (short)0, (c), false, false)

#if defined(__has_builtin)
#if __has_builtin(__builtin_amdgcn_tensor_load_to_lds) && __has_builtin(__builtin_amdgcn_s_wait_tensorcnt)
#define HAVE_TDM 1
#endif
#endif
#ifndef HAVE_TDM
#define HAVE_TDM 0
#endif

// ---------------------------------------------------------------- constants
static constexpr int BB  = 4;
static constexpr int SS  = 1024;
static constexpr int DM  = 1024;
static constexpr int HH  = 16;
static constexpr int DKK = 64;
static constexpr int MROWS = BB * SS;   // 4096

// ---------------------------------------------------------------- f32 -> f16
__global__ void k_cvt_f16(const float* __restrict__ in, _Float16* __restrict__ out, int n) {
  int i = blockIdx.x * blockDim.x + threadIdx.x;
  int stride = gridDim.x * blockDim.x;
  for (; i < n; i += stride) out[i] = (_Float16)in[i];
}

#if HAVE_TDM
// Issue one 2D TDM tile load: tile_dim1 rows x tile_dim0 elems (f16) from a
// [tensor_dim1 x tensor_dim0] row-major tensor (row stride = stride0 elems)
// into LDS at byte offset lds_off, with a 4-dword pad after every 16 dwords
// (=> LDS row stride 80 B for a 32-half tile row).  D# per ISA 8.3/8.4.
// Toolchain here is the 6-arg builtin: (v4u, v8i, v4i, v4i, v8i, i32 cpol).
__device__ __forceinline__ void tdm_load_tile_f16(unsigned lds_off, const _Float16* gptr,
                                                  unsigned tensor_dim0, unsigned tensor_dim1,
                                                  unsigned stride0, unsigned tile_dim0,
                                                  unsigned tile_dim1) {
  unsigned long long ga = (unsigned long long)(uintptr_t)gptr;
  v4u g0;
  g0[0] = 1u;                                          // count=1, user mode, no gather
  g0[1] = lds_off;                                     // LDS byte address
  g0[2] = (unsigned)(ga & 0xFFFFFFFFu);                // global_addr[31:0]
  g0[3] = (unsigned)((ga >> 32) & 0x1FFFFFFu) | (2u << 30);  // global_addr[56:32] | type=2
  v8i_t g1;
  g1[0] = (1 << 16)      // data_size = 2 bytes
        | (1 << 20)      // pad_enable
        | (3 << 22)      // pad_interval: every 16 dwords
        | (3 << 25);     // pad_amount: 4 dwords
  g1[1] = (int)((tensor_dim0 & 0xFFFFu) << 16);                      // tensor_dim0 lo
  g1[2] = (int)(((tensor_dim0 >> 16) & 0xFFFFu) | ((tensor_dim1 & 0xFFFFu) << 16));
  g1[3] = (int)(((tensor_dim1 >> 16) & 0xFFFFu) | ((tile_dim0 & 0xFFFFu) << 16));
  g1[4] = (int)(tile_dim1 & 0xFFFFu);                                // tile_dim1, tile_dim2=0
  g1[5] = (int)stride0;                                              // tensor_dim0_stride lo32
  g1[6] = 0;
  g1[7] = 0;
  v4i_t z4 = {0, 0, 0, 0};
  v8i_t z8 = {0, 0, 0, 0, 0, 0, 0, 0};
  __builtin_amdgcn_tensor_load_to_lds(g0, g1, z4, z4, z8, 0);
}
#endif

// ---------------------------------------------------------------- GEMM: out = A @ W^T + bias
// A: [M,K] f16 row-major, W: [N,K] f16 row-major, bias: [N] f32.
// Block: 256 thr = 8 waves; wave tile 16x64 (4 accumulators); block tile 64(M) x 128(N).
// TDM path: double-buffered LDS tiles (A 64x32, B 128x32, padded 80B row stride).
static constexpr unsigned GEMM_ROWB   = 80;                       // padded LDS row bytes
static constexpr unsigned GEMM_A_BYTES = 64 * GEMM_ROWB;          // 5120
static constexpr unsigned GEMM_B_BYTES = 128 * GEMM_ROWB;         // 10240
static constexpr unsigned GEMM_SMEM    = 2 * GEMM_A_BYTES + 2 * GEMM_B_BYTES; // 30720

template <bool OUT16>
__global__ void k_gemm_wmma(const _Float16* __restrict__ A, const _Float16* __restrict__ W,
                            const float* __restrict__ bias, void* __restrict__ outp,
                            int M, int N, int K) {
  const int lane = threadIdx.x & 31;
  const int w    = threadIdx.x >> 5;
  const int wm   = w & 3;          // 0..3
  const int wn   = w >> 2;         // 0..1
  const int m0   = blockIdx.y * 64  + wm * 16;
  const int n0   = blockIdx.x * 128 + wn * 64;
  const int hl   = lane >> 4;
  const int l15  = lane & 15;

  v8f acc[4] = {v8f{}, v8f{}, v8f{}, v8f{}};

#if HAVE_TDM
  extern __shared__ char smem[];
  const unsigned offA[2] = {0u, GEMM_A_BYTES};
  const unsigned offB[2] = {2 * GEMM_A_BYTES, 2 * GEMM_A_BYTES + GEMM_B_BYTES};
  const unsigned ldsBase = (unsigned)(uintptr_t)smem;   // LDS byte offset of dynamic region
  const _Float16* gA = A + (size_t)(blockIdx.y * 64) * K;
  const _Float16* gB = W + (size_t)(blockIdx.x * 128) * K;
  const int nIter = K / 32;

  if (w == 0) {   // prologue: stage buffer 0
    tdm_load_tile_f16(ldsBase + offA[0], gA, K, M, K, 32, 64);
    tdm_load_tile_f16(ldsBase + offB[0], gB, K, N, K, 32, 128);
  }

  for (int it = 0; it < nIter; ++it) {
    const int cur = it & 1;
    if (w == 0) {
      if (it + 1 < nIter) {
        const int k1 = (it + 1) * 32;
        tdm_load_tile_f16(ldsBase + offA[cur ^ 1], gA + k1, K, M, K, 32, 64);
        tdm_load_tile_f16(ldsBase + offB[cur ^ 1], gB + k1, K, N, K, 32, 128);
        __builtin_amdgcn_s_wait_tensorcnt(2);   // current buffer's 2 loads done
      } else {
        __builtin_amdgcn_s_wait_tensorcnt(0);
      }
    }
    __syncthreads();                            // buffer `cur` visible to all waves

    // A fragment from LDS: local row = wm*16 + l15, halves {hl*8+0..7, hl*8+16..23}
    const char* aRow = smem + offA[cur] + (unsigned)(wm * 16 + l15) * GEMM_ROWB + hl * 16;
    v8h alo = *(const v8h*)(aRow);
    v8h ahi = *(const v8h*)(aRow + 32);
    v16h af;
#pragma unroll
    for (int i = 0; i < 8; ++i) { af[i] = alo[i]; af[8 + i] = ahi[i]; }

#pragma unroll
    for (int nt = 0; nt < 4; ++nt) {
      const char* bRow = smem + offB[cur] +
                         (unsigned)(wn * 64 + nt * 16 + l15) * GEMM_ROWB + hl * 16;
      v8h blo = *(const v8h*)(bRow);
      v8h bhi = *(const v8h*)(bRow + 32);
      v16h bf;
#pragma unroll
      for (int i = 0; i < 8; ++i) { bf[i] = blo[i]; bf[8 + i] = bhi[i]; }
      acc[nt] = WMMA16(af, bf, acc[nt]);
    }
    __syncthreads();                            // all reads of `cur` done before overwrite
  }
#else
  const size_t arow = (size_t)(m0 + l15) * K;
  for (int k0 = 0; k0 < K; k0 += 32) {
    const _Float16* ap = A + arow + k0 + hl * 8;
    v8h alo = *(const v8h*)(ap);
    v8h ahi = *(const v8h*)(ap + 16);
    v16h af;
#pragma unroll
    for (int i = 0; i < 8; ++i) { af[i] = alo[i]; af[8 + i] = ahi[i]; }
#pragma unroll
    for (int nt = 0; nt < 4; ++nt) {
      const _Float16* bp = W + (size_t)(n0 + nt * 16 + l15) * K + k0 + hl * 8;
      v8h blo = *(const v8h*)(bp);
      v8h bhi = *(const v8h*)(bp + 16);
      v16h bf;
#pragma unroll
      for (int i = 0; i < 8; ++i) { bf[i] = blo[i]; bf[8 + i] = bhi[i]; }
      acc[nt] = WMMA16(af, bf, acc[nt]);
    }
  }
#endif

#pragma unroll
  for (int nt = 0; nt < 4; ++nt) {
#pragma unroll
    for (int r = 0; r < 8; ++r) {
      const int m = m0 + r + hl * 8;
      const int n = n0 + nt * 16 + l15;
      const float vv = acc[nt][r] + bias[n];
      if (OUT16) ((_Float16*)outp)[(size_t)m * N + n] = (_Float16)vv;
      else       ((float*)outp)[(size_t)m * N + n]    = vv;
    }
  }
}

// ---------------------------------------------------------------- attention with distance decay
static constexpr int SST = 1028;   // f32 stride (pad vs 64 banks)
static constexpr int AST = 1040;   // f16 stride
static constexpr size_t SMEM_ATTN =
    (size_t)16 * SST * 4 + (size_t)8 * 1024 * 4 + (size_t)16 * AST * 2 + 32;

__global__ void k_attn_decay(const _Float16* __restrict__ Qp, const _Float16* __restrict__ Kp,
                             const _Float16* __restrict__ Vp, const float* __restrict__ gammas,
                             _Float16* __restrict__ Cout) {
  extern __shared__ char smem[];
  float*    sS   = (float*)smem;                         // 16*SST
  float*    pbuf = sS + 16 * SST;                        // 8*1024
  _Float16* aT   = (_Float16*)(pbuf + 8 * 1024);         // 16*AST
  float*    sGam = (float*)(aT + 16 * AST);

  const int qt = blockIdx.x, h = blockIdx.y, b = blockIdx.z;
  const int q0 = qt * 16;
  const int lane = threadIdx.x & 31;
  const int w    = threadIdx.x >> 5;
  const int hl   = lane >> 4;
  const int l15  = lane & 15;

  if (threadIdx.x == 0) {
    float g  = gammas[h];
    float sp = (g > 20.f) ? g : log1pf(expf(g));          // softplus
    *sGam = -sp;
  }

  // ---- Phase 1: scores[16][1024] = Q(16x64) @ K^T / 8 -------------------
  const _Float16* qrow = Qp + (size_t)(b * SS + q0 + l15) * DM + h * DKK;
  v16h aq0, aq1;
  {
    const _Float16* p0 = qrow + 0  + hl * 8;
    const _Float16* p1 = qrow + 32 + hl * 8;
    v8h x0 = *(const v8h*)p0, x1 = *(const v8h*)(p0 + 16);
    v8h y0 = *(const v8h*)p1, y1 = *(const v8h*)(p1 + 16);
#pragma unroll
    for (int i = 0; i < 8; ++i) {
      aq0[i] = x0[i]; aq0[8 + i] = x1[i];
      aq1[i] = y0[i]; aq1[8 + i] = y1[i];
    }
  }
  for (int kt = w * 8; kt < w * 8 + 8; ++kt) {            // 64 key tiles / 8 waves
    v8f acc = {};
    const _Float16* krow = Kp + (size_t)(b * SS + kt * 16 + l15) * DM + h * DKK;
#pragma unroll
    for (int kk = 0; kk < 2; ++kk) {
      const _Float16* bp = krow + kk * 32 + hl * 8;
      v8h b0 = *(const v8h*)bp, b1 = *(const v8h*)(bp + 16);
      v16h bf;
#pragma unroll
      for (int i = 0; i < 8; ++i) { bf[i] = b0[i]; bf[8 + i] = b1[i]; }
      acc = WMMA16(kk == 0 ? aq0 : aq1, bf, acc);
    }
#pragma unroll
    for (int r = 0; r < 8; ++r)
      sS[(r + hl * 8) * SST + kt * 16 + l15] = acc[r] * 0.125f;
  }
  __syncthreads();

  // ---- Phase 2: decay + double softmax, 2 rows per wave -----------------
  const float gamma = *sGam;
  for (int rr = 0; rr < 2; ++rr) {
    const int row = w + rr * 8;
    const int iq  = q0 + row;
    float* srow = sS + row * SST;
    float* prow = pbuf + w * 1024;
    const int c0 = lane * 32;

    float m1 = -3.4e38f;
    for (int i = 0; i < 32; ++i) m1 = fmaxf(m1, srow[c0 + i]);
#pragma unroll
    for (int o = 16; o > 0; o >>= 1) m1 = fmaxf(m1, __shfl_xor(m1, o, 32));
    float s1 = 0.f;
    for (int i = 0; i < 32; ++i) s1 += expf(srow[c0 + i] - m1);
#pragma unroll
    for (int o = 16; o > 0; o >>= 1) s1 += __shfl_xor(s1, o, 32);
    const float inv1 = 1.0f / s1;

    float run = 0.f;
    for (int i = 0; i < 32; ++i) {
      const int c = c0 + i;
      float p = expf(srow[c] - m1) * inv1;
      if (c > iq) p = 0.f;
      run += p;
      prow[c] = run;
    }
    float incl = run;
#pragma unroll
    for (int d = 1; d < 32; d <<= 1) {
      float t = __shfl_up(incl, d, 32);
      if (lane >= d) incl += t;
    }
    const float off       = incl - run;
    const float disttotal = __shfl(incl, 31, 32);

    float m2 = -3.4e38f;
    for (int i = 0; i < 32; ++i) {
      const int c = c0 + i;
      const float distcum = off + prow[c];
      const float pe  = fabsf((float)(iq - c));
      const float x   = (disttotal - distcum) * pe;
      const float dsc = sqrtf(fmaxf(x, 0.f));
      float eff = expf(dsc * gamma);
      eff = fminf(fmaxf(eff, 1e-5f), 1e5f);
      const float ns = (c <= iq) ? srow[c] * eff : -1e32f;
      srow[c] = ns;
      m2 = fmaxf(m2, ns);
    }
#pragma unroll
    for (int o = 16; o > 0; o >>= 1) m2 = fmaxf(m2, __shfl_xor(m2, o, 32));
    float s2 = 0.f;
    for (int i = 0; i < 32; ++i) s2 += expf(srow[c0 + i] - m2);
#pragma unroll
    for (int o = 16; o > 0; o >>= 1) s2 += __shfl_xor(s2, o, 32);
    const float inv2 = 1.0f / s2;

    _Float16* arow = aT + row * AST;
    for (int i = 0; i < 32; ++i) {
      const int c = c0 + i;
      const float p = (c <= iq) ? expf(srow[c] - m2) * inv2 : 0.f;
      arow[c] = (_Float16)p;
    }
  }
  __syncthreads();

  // ---- Phase 3: out(16x64) = attn(16x1024) @ V(1024x64), 4 waves --------
  if (w < 4) {
    const int n0 = w * 16;
    const int d  = h * DKK + n0 + l15;
    v8f acc = {};
    for (int kt = 0; kt < 32; ++kt) {
      if (kt + 1 < 32)   // prefetch next V tile rows -> global_prefetch_b8
        __builtin_prefetch(Vp + (size_t)(b * SS + (kt + 1) * 32 + hl * 16) * DM + d, 0, 1);

      const _Float16* ap = aT + (size_t)l15 * AST + kt * 32 + hl * 16;   // LDS
      v8h a0 = *(const v8h*)ap, a1 = *(const v8h*)(ap + 8);
      v16h af;
#pragma unroll
      for (int i = 0; i < 8; ++i) { af[i] = a0[i]; af[8 + i] = a1[i]; }

      const int kbase = kt * 32 + hl * 16;
      v16h bf;
#pragma unroll
      for (int i = 0; i < 16; ++i)
        bf[i] = Vp[(size_t)(b * SS + kbase + i) * DM + d];
      acc = WMMA16(af, bf, acc);
    }
#pragma unroll
    for (int r = 0; r < 8; ++r) {
      const int m = q0 + r + hl * 8;
      Cout[(size_t)(b * SS + m) * DM + h * DKK + n0 + l15] = (_Float16)acc[r];
    }
  }
}

// ---------------------------------------------------------------- launcher
extern "C" void kernel_launch(void* const* d_in, const int* in_sizes, int n_in,
                              void* d_out, int out_size, void* d_ws, size_t ws_size,
                              hipStream_t stream) {
  (void)in_sizes; (void)n_in; (void)out_size; (void)ws_size;

  const float* q   = (const float*)d_in[0];
  const float* k   = (const float*)d_in[1];
  const float* v   = (const float*)d_in[2];
  const float* Wq  = (const float*)d_in[3];
  const float* bq  = (const float*)d_in[4];
  const float* Wk  = (const float*)d_in[5];
  const float* bk  = (const float*)d_in[6];
  const float* Wv  = (const float*)d_in[7];
  const float* bv  = (const float*)d_in[8];
  const float* Wo  = (const float*)d_in[9];
  const float* bo  = (const float*)d_in[10];
  const float* gam = (const float*)d_in[11];
  float* out = (float*)d_out;

  char* ws = (char*)d_ws;
  size_t off = 0;
  auto carve = [&](size_t bytes) -> char* {
    char* p = ws + off;
    off += (bytes + 255) & ~(size_t)255;
    return p;
  };
  const size_t ACT = (size_t)MROWS * DM;   // 4M elems
  const size_t WEL = (size_t)DM * DM;      // 1M elems

  _Float16* q16  = (_Float16*)carve(ACT * 2);
  _Float16* k16  = (_Float16*)carve(ACT * 2);
  _Float16* v16  = (_Float16*)carve(ACT * 2);
  _Float16* Wq16 = (_Float16*)carve(WEL * 2);
  _Float16* Wk16 = (_Float16*)carve(WEL * 2);
  _Float16* Wv16 = (_Float16*)carve(WEL * 2);
  _Float16* Wo16 = (_Float16*)carve(WEL * 2);
  _Float16* Qp16 = (_Float16*)carve(ACT * 2);
  _Float16* Kp16 = (_Float16*)carve(ACT * 2);
  _Float16* Vp16 = (_Float16*)carve(ACT * 2);
  _Float16* c16  = (_Float16*)carve(ACT * 2);

  // 1) convert everything to f16
  k_cvt_f16<<<2048, 256, 0, stream>>>(q,  q16,  (int)ACT);
  k_cvt_f16<<<2048, 256, 0, stream>>>(k,  k16,  (int)ACT);
  k_cvt_f16<<<2048, 256, 0, stream>>>(v,  v16,  (int)ACT);
  k_cvt_f16<<<1024, 256, 0, stream>>>(Wq, Wq16, (int)WEL);
  k_cvt_f16<<<1024, 256, 0, stream>>>(Wk, Wk16, (int)WEL);
  k_cvt_f16<<<1024, 256, 0, stream>>>(Wv, Wv16, (int)WEL);
  k_cvt_f16<<<1024, 256, 0, stream>>>(Wo, Wo16, (int)WEL);

  // 2) Q/K/V projections (f16 out), TDM double-buffered LDS staging
  const dim3 ggrid(DM / 128, MROWS / 64);
  k_gemm_wmma<true><<<ggrid, 256, GEMM_SMEM, stream>>>(q16, Wq16, bq, Qp16, MROWS, DM, DM);
  k_gemm_wmma<true><<<ggrid, 256, GEMM_SMEM, stream>>>(k16, Wk16, bk, Kp16, MROWS, DM, DM);
  k_gemm_wmma<true><<<ggrid, 256, GEMM_SMEM, stream>>>(v16, Wv16, bv, Vp16, MROWS, DM, DM);

  // 3) distance-decay attention -> concat f16
  k_attn_decay<<<dim3(SS / 16, HH, BB), 256, SMEM_ATTN, stream>>>(Qp16, Kp16, Vp16, gam, c16);

  // 4) output projection -> f32 result
  k_gemm_wmma<false><<<ggrid, 256, GEMM_SMEM, stream>>>(c16, Wo16, bo, out, MROWS, DM, DM);
}
